// T5EncoderSelfAttentionFlex_13451837571869
// MI455X (gfx1250) — matrix operand order repaired and checked by reading, not compiled, not run on hardware
//
#include <hip/hip_runtime.h>

// ---------------- problem constants ----------------
#define S      2048
#define BATCH  2
#define NHEAD  16
#define HD     64
#define HIDDEN 1024
#define QKVN   (3 * NHEAD * HD)     // 3072
#define BIAS_W (2 * S - 1)          // 4095
#define BPAD   40                   // padded LDS row stride (elements) for 32-elem rows

typedef __bf16 bf16_t;
typedef __attribute__((ext_vector_type(16))) __bf16 v16bf;
typedef __attribute__((ext_vector_type(8)))  __bf16 v8bf;
typedef __attribute__((ext_vector_type(8)))  float  v8f;

// vector types used by gfx1250 memory builtins
typedef int          vsi4 __attribute__((vector_size(16)));
typedef unsigned int vu4  __attribute__((vector_size(16)));
typedef int          vi4  __attribute__((vector_size(16)));
typedef int          vi8  __attribute__((vector_size(32)));
typedef __attribute__((address_space(1))) vsi4*   gvec4_p;
typedef __attribute__((address_space(3))) vsi4*   lvec4_p;
typedef __attribute__((address_space(3))) __bf16* as3_bf16_p;

// gfx1250 async global->LDS path (guarded; falls back to load + ds_store)
#if defined(__has_builtin)
#if __has_builtin(__builtin_amdgcn_global_load_async_to_lds_b128) && \
    __has_builtin(__builtin_amdgcn_s_wait_asynccnt)
#define HAVE_ASYNC_LDS 1
#endif
#if __has_builtin(__builtin_amdgcn_tensor_load_to_lds) && \
    __has_builtin(__builtin_amdgcn_s_wait_tensorcnt)
#define HAVE_TDM 1
#endif
#endif

__device__ __forceinline__ void stage16(const bf16_t* g, bf16_t* l) {
#ifdef HAVE_ASYNC_LDS
  __builtin_amdgcn_global_load_async_to_lds_b128((gvec4_p)(g), (lvec4_p)(l),
                                                 0, 0);
#else
  *(v8bf*)l = *(const v8bf*)g;
#endif
}
__device__ __forceinline__ void wait_stage() {
#ifdef HAVE_ASYNC_LDS
  __builtin_amdgcn_s_wait_asynccnt(0);
#endif
}

// ---- Tensor Data Mover: load a 64(rows) x 32(elems) bf16 tile into LDS ----
// D# per CDNA5 ISA ch.8: data_size=2B; pad_enable with pad_interval=16 DWORDs
// (one 32-elem row) and pad_amount=4 DWORDs -> LDS rows land at BPAD=40 elems.
__device__ __forceinline__ void tdm_load_tile_64x32(const bf16_t* gtile,
                                                    unsigned lds_off,
                                                    int ld, int nrows) {
#ifdef HAVE_TDM
  unsigned long long ga = (unsigned long long)gtile;
  vu4 g0;
  g0[0] = 1u;                                      // count=1, no gather
  g0[1] = lds_off;                                 // lds_addr (bytes)
  g0[2] = (unsigned)ga;                            // global_addr[31:0]
  g0[3] = (unsigned)((ga >> 32) & 0x01FFFFFFu)     // global_addr[56:32]
        | 0x80000000u;                             // type=2 ("image")
  vi8 g1;
  g1[0] = (1 << 16)                                // data_size = 2 bytes
        | (1 << 20)                                // pad_enable
        | (3 << 22)                                // pad_interval: 16 DWORDs
        | (3 << 25);                               // pad_amount: 4 DWORDs
  g1[1] = (int)((unsigned)(ld & 0xFFFF) << 16);    // tensor_dim0[15:0]
  g1[2] = (int)(((unsigned)ld >> 16)               // tensor_dim0[31:16]
        | ((unsigned)(nrows & 0xFFFF) << 16));     // tensor_dim1[15:0]
  g1[3] = (int)(((unsigned)nrows >> 16)            // tensor_dim1[31:16]
        | (32u << 16));                            // tile_dim0 = 32 elems
  g1[4] = 64;                                      // tile_dim1 = 64, tile_dim2=0
  g1[5] = ld;                                      // tensor_dim0_stride[31:0]
  g1[6] = 0;
  g1[7] = 0;
  vi4 z4 = {0, 0, 0, 0};
#if __clang_major__ >= 23
  vi8 z8 = {0, 0, 0, 0, 0, 0, 0, 0};
  __builtin_amdgcn_tensor_load_to_lds(g0, g1, z4, z4, z8, 0);
#else
  __builtin_amdgcn_tensor_load_to_lds(g0, g1, z4, z4, 0);
#endif
#endif
}
__device__ __forceinline__ void tdm_wait() {
#ifdef HAVE_TDM
  __builtin_amdgcn_s_wait_tensorcnt(0);
#endif
}

// ---------------- WMMA helpers ----------------
__device__ __forceinline__ v8f wmma_bf16(v16bf a, v16bf b, v8f c) {
  return __builtin_amdgcn_wmma_f32_16x16x32_bf16(false, a, false, b,
                                                 (short)0, c, false, false);
}

// A-matrix 16x32 fragment, row-major source (two 16B chunks per lane).
__device__ __forceinline__ v16bf load_a_frag(const bf16_t* base, int ld) {
  int lane = threadIdx.x & 31;
  const bf16_t* p = base + (lane & 15) * ld + ((lane >> 4) << 4);
  v8bf lo = *(const v8bf*)(p);
  v8bf hi = *(const v8bf*)(p + 8);
  v16bf r;
#pragma unroll
  for (int i = 0; i < 8; ++i) { r[i] = lo[i]; r[8 + i] = hi[i]; }
  return r;
}

// B-matrix 32x16 fragment where B = W^T chunk; base -> W[n0][k0]. 32B/lane.
__device__ __forceinline__ v16bf load_b_frag(const bf16_t* base, int ld) {
  int lane = threadIdx.x & 31;
  const bf16_t* p = base + (lane & 15) * ld + ((lane >> 4) << 4);
  return *(const v16bf*)p;
}

// Same, from LDS with padded rows (16B-aligned halves).
__device__ __forceinline__ v16bf load_b_frag_lds(const bf16_t* base, int ld) {
  int lane = threadIdx.x & 31;
  const bf16_t* p = base + (lane & 15) * ld + ((lane >> 4) << 4);
  v8bf lo = *(const v8bf*)(p);
  v8bf hi = *(const v8bf*)(p + 8);
  v16bf r;
#pragma unroll
  for (int i = 0; i < 8; ++i) { r[i] = lo[i]; r[8 + i] = hi[i]; }
  return r;
}

__device__ __forceinline__ float rmax16(float v) {
#pragma unroll
  for (int o = 8; o >= 1; o >>= 1) v = fmaxf(v, __shfl_xor(v, o, 16));
  return v;
}
__device__ __forceinline__ float rsum16(float v) {
#pragma unroll
  for (int o = 8; o >= 1; o >>= 1) v += __shfl_xor(v, o, 16);
  return v;
}

// ---------------- kernel 1: fp32 -> bf16 ----------------
__global__ void cvt_bf16_kernel(const float* __restrict__ in,
                                bf16_t* __restrict__ out, int n) {
  int i = blockIdx.x * blockDim.x + threadIdx.x;
  int stride = gridDim.x * blockDim.x;
  for (; i < n; i += stride) out[i] = (bf16_t)in[i];
}

// ---------------- kernel 2: T5 relative bias table ----------------
__global__ void bias_table_kernel(const float* __restrict__ emb,
                                  float* __restrict__ tab) {
  int idx = blockIdx.x * blockDim.x + threadIdx.x;
  int stride = gridDim.x * blockDim.x;
  const float coeff = 2.885390082f;  // 8 / ln(128/8)
  for (; idx < NHEAD * BIAS_W; idx += stride) {
    int h = idx / BIAS_W;
    int p = idx - h * BIAS_W;
    int rel = p - (S - 1);
    int a = rel < 0 ? -rel : rel;
    int base = rel > 0 ? 16 : 0;
    int am = a > 1 ? a : 1;
    int distant = 8 + (int)(__logf((float)am * 0.125f) * coeff);
    distant = distant < 15 ? distant : 15;
    int bucket = base + (a < 8 ? a : distant);
    tab[h * BIAS_W + p] = emb[bucket * NHEAD + h];
  }
}

// ------- GEMM core A: 32x64/wave, async-load staged B (double buffer) ------
template <int KD>
__device__ __forceinline__ void gemm_core_32x64(const bf16_t* __restrict__ arow,
                                                const bf16_t* __restrict__ wbase,
                                                bf16_t* bs, v8f acc[8]) {
  const int tid = threadIdx.x;
  const int ln = tid >> 2;            // 0..63: row within B tile
  const int lk = (tid & 3) << 3;      // 0,8,16,24: k chunk (16B)
  const bf16_t* gsrc = wbase + (size_t)ln * KD + lk;
  bf16_t* ldst = bs + ln * BPAD + lk;
  const int NK = KD / 32;

  stage16(gsrc, ldst);                        // tile 0 -> buf 0
  v16bf a0 = load_a_frag(arow, KD);
  v16bf a1 = load_a_frag(arow + (size_t)16 * KD, KD);

  for (int i = 0; i < NK; ++i) {
    wait_stage();
    __syncthreads();
    const bf16_t* buf = bs + (i & 1) * (64 * BPAD);
    v16bf na0 = a0, na1 = a1;
    if (i + 1 < NK) {
      stage16(gsrc + (i + 1) * 32, ldst + ((i + 1) & 1) * (64 * BPAD));
      na0 = load_a_frag(arow + (i + 1) * 32, KD);
      na1 = load_a_frag(arow + (size_t)16 * KD + (i + 1) * 32, KD);
    }
#pragma unroll
    for (int t = 0; t < 4; ++t) {
      v16bf b = load_b_frag_lds(buf + t * 16 * BPAD, BPAD);
      acc[t]     = wmma_bf16(a0, b, acc[t]);
      acc[4 + t] = wmma_bf16(a1, b, acc[4 + t]);
    }
    a0 = na0; a1 = na1;
  }
}

// ------- GEMM core B: 32x64/wave, TDM-staged B (one DMA per K-step) --------
template <int KD>
__device__ __forceinline__ void gemm_core_32x64_tdm(
    const bf16_t* __restrict__ arow, const bf16_t* __restrict__ wbase,
    bf16_t* bs, unsigned bs_off, v8f acc[8]) {
  const int NK = KD / 32;
  const bool issuer = (threadIdx.x < 32);    // wave 0 drives the TDM

  if (issuer) tdm_load_tile_64x32(wbase, bs_off, KD, 64);  // tile 0 -> buf 0
  v16bf a0 = load_a_frag(arow, KD);
  v16bf a1 = load_a_frag(arow + (size_t)16 * KD, KD);

  for (int i = 0; i < NK; ++i) {
    if (issuer) tdm_wait();                  // tile i landed in LDS
    __syncthreads();
    const bf16_t* buf = bs + (i & 1) * (64 * BPAD);
    v16bf na0 = a0, na1 = a1;
    if (i + 1 < NK) {
      if (issuer)
        tdm_load_tile_64x32(wbase + (i + 1) * 32,
                            bs_off + ((i + 1) & 1) * (64 * BPAD * 2), KD, 64);
      na0 = load_a_frag(arow + (i + 1) * 32, KD);
      na1 = load_a_frag(arow + (size_t)16 * KD + (i + 1) * 32, KD);
    }
#pragma unroll
    for (int t = 0; t < 4; ++t) {
      v16bf b = load_b_frag_lds(buf + t * 16 * BPAD, BPAD);
      acc[t]     = wmma_bf16(a0, b, acc[t]);
      acc[4 + t] = wmma_bf16(a1, b, acc[4 + t]);
    }
    a0 = na0; a1 = na1;
  }
}

// ---------------- kernel 3: QKV GEMM + scatter (async-staged B) ------------
__global__ __launch_bounds__(256) void qkv_gemm_kernel(
    const bf16_t* __restrict__ xb, const bf16_t* __restrict__ w,
    bf16_t* __restrict__ Q, bf16_t* __restrict__ K, bf16_t* __restrict__ Vt) {
  __shared__ __attribute__((aligned(32))) bf16_t Bs[2 * 64 * BPAD];
  int wave = threadIdx.x >> 5;
  int m0 = blockIdx.x * 256 + wave * 32;
  int n0 = blockIdx.y * 64;
  v8f acc[8] = {v8f{}, v8f{}, v8f{}, v8f{}, v8f{}, v8f{}, v8f{}, v8f{}};
  gemm_core_32x64<HIDDEN>(xb + (size_t)m0 * HIDDEN,
                          w + (size_t)n0 * HIDDEN, Bs, acc);

  int lane = threadIdx.x & 31;
  int hhalf = lane >> 4, c = lane & 15;
  int head = (n0 >> 6) & (NHEAD - 1);
  int tsel = n0 >> 10;  // 0=Q, 1=K, 2=V
#pragma unroll
  for (int half = 0; half < 2; ++half) {
#pragma unroll
    for (int t = 0; t < 4; ++t) {
#pragma unroll
      for (int r = 0; r < 8; ++r) {
        int row = m0 + half * 16 + r + hhalf * 8;
        int b = row >> 11, s = row & (S - 1);
        int d = t * 16 + c;
        float v = acc[half * 4 + t][r];
        size_t qk_idx = (((size_t)b * NHEAD + head) * S + s) * HD + d;
        if (tsel == 0)      Q[qk_idx] = (bf16_t)(v * 0.125f);  // fold hd^-0.5
        else if (tsel == 1) K[qk_idx] = (bf16_t)v;
        else Vt[(((size_t)b * NHEAD + head) * HD + d) * S + s] = (bf16_t)v;
      }
    }
  }
}

// ---------------- kernel 4: flash attention ----------------
__global__ __launch_bounds__(128) void attn_kernel(
    const bf16_t* __restrict__ Q, const bf16_t* __restrict__ K,
    const bf16_t* __restrict__ Vt, const float* __restrict__ bias,
    bf16_t* __restrict__ A) {
  __shared__ __attribute__((aligned(32))) bf16_t Plds[4][16 * BPAD];
  int wave = threadIdx.x >> 5;
  int lane = threadIdx.x & 31;
  int hhalf = lane >> 4, c = lane & 15;

  int bid = blockIdx.x;                 // [b][h][qt]
  int qt = bid & 31;
  int h  = (bid >> 5) & (NHEAD - 1);
  int b  = bid >> 9;
  int q0 = qt * 64 + wave * 16;

  const bf16_t* Qb = Q  + ((size_t)b * NHEAD + h) * S * HD;
  const bf16_t* Kb = K  + ((size_t)b * NHEAD + h) * S * HD;
  const bf16_t* Vb = Vt + ((size_t)b * NHEAD + h) * HD * S;
  const float*  bt = bias + (size_t)h * BIAS_W;
  bf16_t* P = &Plds[wave][0];

  v16bf aq0 = load_a_frag(Qb + (size_t)q0 * HD, HD);
  v16bf aq1 = load_a_frag(Qb + (size_t)q0 * HD + 32, HD);

  v8f o[4] = {v8f{}, v8f{}, v8f{}, v8f{}};
  float m[8], l[8];
#pragma unroll
  for (int r = 0; r < 8; ++r) { m[r] = -1e30f; l[r] = 0.0f; }

  v16bf bk0 = load_b_frag(Kb, HD);
  v16bf bk1 = load_b_frag(Kb + 32, HD);
  v16bf bk2 = load_b_frag(Kb + (size_t)16 * HD, HD);
  v16bf bk3 = load_b_frag(Kb + (size_t)16 * HD + 32, HD);

  for (int kt = 0; kt < S; kt += 32) {
    v8f s0 = v8f{}, s1 = v8f{};
    s0 = wmma_bf16(aq0, bk0, s0);
    s0 = wmma_bf16(aq1, bk1, s0);
    s1 = wmma_bf16(aq0, bk2, s1);
    s1 = wmma_bf16(aq1, bk3, s1);

    // V fragments: independent of softmax -> issue early
    v16bf bv0 = load_b_frag(Vb + kt, S);
    v16bf bv1 = load_b_frag(Vb + (size_t)16 * S + kt, S);
    v16bf bv2 = load_b_frag(Vb + (size_t)32 * S + kt, S);
    v16bf bv3 = load_b_frag(Vb + (size_t)48 * S + kt, S);

    // register-prefetch next K tile before the softmax VALU block
    int kn = kt + 32;
    if (kn < S) {
      bk0 = load_b_frag(Kb + (size_t)kn * HD, HD);
      bk1 = load_b_frag(Kb + (size_t)kn * HD + 32, HD);
      bk2 = load_b_frag(Kb + (size_t)(kn + 16) * HD, HD);
      bk3 = load_b_frag(Kb + (size_t)(kn + 16) * HD + 32, HD);
      __builtin_prefetch(Kb + (size_t)(kn + 32) * HD, 0, 0);
      __builtin_prefetch(Vb + kn + 32, 0, 0);
    }

#pragma unroll
    for (int r = 0; r < 8; ++r) {
      int qrow = q0 + r + hhalf * 8;
      s0[r] += bt[kt + c - qrow + (S - 1)];
      s1[r] += bt[kt + 16 + c - qrow + (S - 1)];
    }

#pragma unroll
    for (int r = 0; r < 8; ++r) {
      float tm = rmax16(fmaxf(s0[r], s1[r]));
      float nm = fmaxf(m[r], tm);
      float al = __expf(m[r] - nm);
      float p0 = __expf(s0[r] - nm);
      float p1 = __expf(s1[r] - nm);
      l[r] = l[r] * al + rsum16(p0 + p1);
      m[r] = nm;
      o[0][r] *= al; o[1][r] *= al; o[2][r] *= al; o[3][r] *= al;
      int mr = r + hhalf * 8;
      P[mr * BPAD + c]      = (bf16_t)p0;
      P[mr * BPAD + 16 + c] = (bf16_t)p1;
    }

    v16bf ap = load_a_frag(P, BPAD);
    o[0] = wmma_bf16(ap, bv0, o[0]);
    o[1] = wmma_bf16(ap, bv1, o[1]);
    o[2] = wmma_bf16(ap, bv2, o[2]);
    o[3] = wmma_bf16(ap, bv3, o[3]);
  }

#pragma unroll
  for (int t = 0; t < 4; ++t) {
#pragma unroll
    for (int r = 0; r < 8; ++r) {
      int qrow = q0 + r + hhalf * 8;
      float v = o[t][r] / l[r];
      A[((size_t)b * S + qrow) * HIDDEN + h * HD + t * 16 + c] = (bf16_t)v;
    }
  }
}

// ---------------- kernel 5: output GEMM (TDM-staged B, fp32 out) -----------
__global__ __launch_bounds__(256) void o_gemm_kernel(
    const bf16_t* __restrict__ A, const bf16_t* __restrict__ w,
    float* __restrict__ out) {
  __shared__ __attribute__((aligned(32))) bf16_t Bs[2 * 64 * BPAD];
  int wave = threadIdx.x >> 5;
  int m0 = blockIdx.x * 256 + wave * 32;
  int n0 = blockIdx.y * 64;
  v8f acc[8] = {v8f{}, v8f{}, v8f{}, v8f{}, v8f{}, v8f{}, v8f{}, v8f{}};
#ifdef HAVE_TDM
  unsigned bs_off = (unsigned)(unsigned long long)(as3_bf16_p)(&Bs[0]);
  gemm_core_32x64_tdm<HIDDEN>(A + (size_t)m0 * HIDDEN,
                              w + (size_t)n0 * HIDDEN, Bs, bs_off, acc);
#else
  gemm_core_32x64<HIDDEN>(A + (size_t)m0 * HIDDEN,
                          w + (size_t)n0 * HIDDEN, Bs, acc);
#endif

  int lane = threadIdx.x & 31;
  int hhalf = lane >> 4, c = lane & 15;
#pragma unroll
  for (int half = 0; half < 2; ++half) {
#pragma unroll
    for (int t = 0; t < 4; ++t) {
#pragma unroll
      for (int r = 0; r < 8; ++r) {
        int row = m0 + half * 16 + r + hhalf * 8;
        out[(size_t)row * HIDDEN + n0 + t * 16 + c] = acc[half * 4 + t][r];
      }
    }
  }
}

// ---------------- launcher ----------------
extern "C" void kernel_launch(void* const* d_in, const int* in_sizes, int n_in,
                              void* d_out, int out_size, void* d_ws, size_t ws_size,
                              hipStream_t stream) {
  const float* x     = (const float*)d_in[0];
  const float* emb   = (const float*)d_in[1];
  const float* qkv_w = (const float*)d_in[2];
  const float* o_w   = (const float*)d_in[3];
  float* out = (float*)d_out;

  char* ws = (char*)d_ws;
  size_t off = 0;
  auto carve = [&](size_t bytes) -> void* {
    void* p = ws + off;
    off += (bytes + 255) & ~(size_t)255;
    return p;
  };
  const size_t n_x   = (size_t)BATCH * S * HIDDEN;
  const size_t n_qkv = (size_t)QKVN * HIDDEN;
  const size_t n_ow  = (size_t)HIDDEN * HIDDEN;
  const size_t n_hd  = (size_t)BATCH * NHEAD * S * HD;

  bf16_t* xb   = (bf16_t*)carve(n_x * 2);
  bf16_t* wqkv = (bf16_t*)carve(n_qkv * 2);
  bf16_t* wo   = (bf16_t*)carve(n_ow * 2);
  bf16_t* Qb   = (bf16_t*)carve(n_hd * 2);
  bf16_t* Kb   = (bf16_t*)carve(n_hd * 2);
  bf16_t* Vt   = (bf16_t*)carve(n_hd * 2);
  bf16_t* Ab   = (bf16_t*)carve(n_x * 2);
  float*  tab  = (float*)carve((size_t)NHEAD * BIAS_W * 4);

  cvt_bf16_kernel<<<2048, 256, 0, stream>>>(x, xb, (int)n_x);
  cvt_bf16_kernel<<<2048, 256, 0, stream>>>(qkv_w, wqkv, (int)n_qkv);
  cvt_bf16_kernel<<<1024, 256, 0, stream>>>(o_w, wo, (int)n_ow);
  bias_table_kernel<<<64, 256, 0, stream>>>(emb, tab);

  // QKV GEMM: M=4096 (256/block), N=3072 (64/block)
  qkv_gemm_kernel<<<dim3(16, 48), 256, 0, stream>>>(xb, wqkv, Qb, Kb, Vt);

  // flash attention: B*H*(S/64) blocks of 4 waves
  attn_kernel<<<BATCH * NHEAD * (S / 64), 128, 0, stream>>>(Qb, Kb, Vt, tab, Ab);

  // output GEMM: M=4096, N=1024
  o_gemm_kernel<<<dim3(16, 16), 256, 0, stream>>>(Ab, wo, out);
}
